// FPCNN_ScanNet_36618891166179
// MI455X (gfx1250) — matrix operand order repaired and checked by reading, not compiled, not run on hardware
//
#include <hip/hip_runtime.h>
#include <hip/hip_bf16.h>

typedef __attribute__((ext_vector_type(16))) _Float16 v16h;
typedef __attribute__((ext_vector_type(8)))  _Float16 v8h;
typedef __attribute__((ext_vector_type(8)))  float    v8f;

// ---------------------------------------------------------------------------
// Split pointcloud [B*N][6] -> xyz f32 [B*N][3], feats f16 [B*N][32] (3 + pad0)
// ---------------------------------------------------------------------------
__global__ void k_split_pc(const float* __restrict__ pc, float* __restrict__ xyz,
                           _Float16* __restrict__ f0, int total) {
  int t = blockIdx.x * blockDim.x + threadIdx.x;
  if (t >= total) return;
  const float* p = pc + (size_t)t * 6;
  xyz[t * 3 + 0] = p[0]; xyz[t * 3 + 1] = p[1]; xyz[t * 3 + 2] = p[2];
  _Float16* o = f0 + (size_t)t * 32;
  o[0] = (_Float16)p[3]; o[1] = (_Float16)p[4]; o[2] = (_Float16)p[5];
#pragma unroll
  for (int c = 3; c < 32; ++c) o[c] = (_Float16)0.f;
}

// ---------------------------------------------------------------------------
// Pack f32 weights [K][N] into WMMA B-matrix per-lane layout, f16, zero-padded.
// k order used here, same as A's (32x16 per k-tile): lane holds col n=lane%16; element e of
// the lane's v16h is K = kt*32 + (lane/16)*16 + e  -> one 32B load per k-tile.
// ---------------------------------------------------------------------------
__global__ void k_pack_w(const float* __restrict__ W, _Float16* __restrict__ out,
                         int K, int Ncol, int ntiles, int total) {
  int t = blockIdx.x * blockDim.x + threadIdx.x;
  if (t >= total) return;
  int e    = t & 15;
  int lane = (t >> 4) & 31;
  int rem  = t >> 9;              // kt*ntiles + nt
  int nt   = rem % ntiles;
  int kt   = rem / ntiles;
  int n    = nt * 16 + (lane & 15);
  int k    = kt * 32 + (lane >> 4) * 16 + e;
  out[t] = (k < K && n < Ncol) ? (_Float16)W[(size_t)k * Ncol + n] : (_Float16)0.f;
}

// centers = xyz[:, ::stride]
__global__ void k_stride_centers(const float* __restrict__ xin, float* __restrict__ ctr,
                                 int total, int M, int Nin, int stride) {
  int t = blockIdx.x * blockDim.x + threadIdx.x;
  if (t >= total) return;
  int b = t / M, m = t - b * M;
  const float* s = xin + ((size_t)b * Nin + (size_t)m * stride) * 3;
  ctr[t * 3 + 0] = s[0]; ctr[t * 3 + 1] = s[1]; ctr[t * 3 + 2] = s[2];
}

// ---------------------------------------------------------------------------
// Ball query: first nsample points (by index order) with d2 < r2, fill with
// first hit — matches reference top_k(-(idx or N)) semantics.
// ---------------------------------------------------------------------------
__global__ void k_ball_query(const float* __restrict__ xyz, const float* __restrict__ ctr,
                             int* __restrict__ idx, int total, int M, int Nin,
                             int nsample, float r2) {
  int t = blockIdx.x * blockDim.x + threadIdx.x;
  if (t >= total) return;
  int b = t / M;
  const float* P = xyz + (size_t)b * Nin * 3;
  float cx = ctr[t * 3 + 0], cy = ctr[t * 3 + 1], cz = ctr[t * 3 + 2];
  int* o = idx + (size_t)t * nsample;
  int cnt = 0, first = 0;
  for (int j = 0; j < Nin && cnt < nsample; ++j) {
    float dx = P[j * 3] - cx, dy = P[j * 3 + 1] - cy, dz = P[j * 3 + 2] - cz;
    float d2 = dx * dx + dy * dy + dz * dz;
    if (d2 < r2) { if (cnt == 0) first = j; o[cnt++] = j; }
  }
  for (; cnt < nsample; ++cnt) o[cnt] = first;
}

// Gather rows: out[r][c] = feats[b*Nin + idx[r]][c]
__global__ void k_gather(const _Float16* __restrict__ f, const int* __restrict__ idx,
                         _Float16* __restrict__ out, int ld, int nperb, int Nin, int total) {
  int t = blockIdx.x * blockDim.x + threadIdx.x;
  if (t >= total) return;
  int r = t / ld, c = t - r * ld;
  int b = r / nperb;
  out[t] = f[((size_t)b * Nin + idx[r]) * ld + c];
}

// ---------------------------------------------------------------------------
// WMMA GEMM: Out[rows x Np] = relu(A[rows x Kp] * W + bias). One wave per
// 16x16 tile. A: row-major f16 (lda=Kp). Wp: pre-packed B layout. f32 accum.
// Padded columns (n >= realN) are written as exact zeros so they can feed the
// next layer's padded-K input untouched.
// ---------------------------------------------------------------------------
__global__ void k_gemm(const _Float16* __restrict__ A, int lda,
                       const _Float16* __restrict__ Wp,
                       const float* __restrict__ bias, int realN,
                       _Float16* __restrict__ out, int ldo,
                       int rows, int Kp, int Np, int relu) {
  int wave = (blockIdx.x * blockDim.x + threadIdx.x) >> 5;
  int lane = threadIdx.x & 31;
  int ntiles = Np >> 4;
  int mt = wave / ntiles;
  int nt = wave - mt * ntiles;
  if (mt * 16 >= rows) return;            // whole wave exits -> EXEC stays all-1s
  int half_ = lane >> 4;
  int mn    = lane & 15;
  // A layout: lane holds row mt*16+mn; per k-tile one contiguous run of 16 K's
  const _Float16* arow = A + (size_t)(mt * 16 + mn) * lda + half_ * 16;
  const _Float16* wb   = Wp + ((size_t)nt * 32 + lane) * 16;
  size_t wstride = (size_t)ntiles * 32 * 16;
  v8f c = {};
  int ktiles = Kp >> 5;
  for (int kt = 0; kt < ktiles; ++kt) {
    v8h alo = *(const v8h*)(arow + kt * 32);
    v8h ahi = *(const v8h*)(arow + kt * 32 + 8);
    v16h a;
#pragma unroll
    for (int i = 0; i < 8; ++i) { a[i] = alo[i]; a[8 + i] = ahi[i]; }
    v16h b = *(const v16h*)(wb + (size_t)kt * wstride);
    if (kt + 1 < ktiles) __builtin_prefetch(arow + (kt + 1) * 32, 0, 1);
    c = __builtin_amdgcn_wmma_f32_16x16x32_f16(false, a, false, b,
                                               (short)0, c, false, false);
  }
  int n = nt * 16 + mn;
  float bv = (n < realN) ? bias[n] : 0.f;
#pragma unroll
  for (int r = 0; r < 8; ++r) {
    int m = mt * 16 + r + half_ * 8;      // C/D layout: lanes 16-31 hold M=8..15
    float v = c[r] + bv;
    if (relu) v = fmaxf(v, 0.f);
    if (n >= realN) v = 0.f;
    out[(size_t)m * ldo + n] = (_Float16)v;
  }
}

// Max over nsample group rows per center/channel
__global__ void k_maxpool(const _Float16* __restrict__ h, _Float16* __restrict__ out,
                          int ld, int realC, int nsample, int total) {
  int t = blockIdx.x * blockDim.x + threadIdx.x;
  if (t >= total) return;
  int r = t / ld, c = t - r * ld;
  if (c >= realC) { out[t] = (_Float16)0.f; return; }
  const _Float16* base = h + (size_t)r * nsample * ld + c;
  float m = -3.0e38f;
  for (int s = 0; s < nsample; ++s) m = fmaxf(m, (float)base[(size_t)s * ld]);
  out[t] = (_Float16)m;
}

// 3-nearest-neighbor indices + inverse-distance weights (matches reference)
__global__ void k_knn3(const float* __restrict__ x1, const float* __restrict__ x2,
                       int* __restrict__ idx3, float* __restrict__ w3,
                       int total, int n1, int n2) {
  int t = blockIdx.x * blockDim.x + threadIdx.x;
  if (t >= total) return;
  int b = t / n1;
  const float* P = x2 + (size_t)b * n2 * 3;
  float px = x1[t * 3], py = x1[t * 3 + 1], pz = x1[t * 3 + 2];
  float d0 = 3e38f, d1 = 3e38f, d2 = 3e38f;
  int i0 = 0, i1 = 0, i2 = 0;
  for (int j = 0; j < n2; ++j) {
    float dx = P[j * 3] - px, dy = P[j * 3 + 1] - py, dz = P[j * 3 + 2] - pz;
    float d = dx * dx + dy * dy + dz * dz;
    if (d < d0)      { d2 = d1; i2 = i1; d1 = d0; i1 = i0; d0 = d; i0 = j; }
    else if (d < d1) { d2 = d1; i2 = i1; d1 = d;  i1 = j; }
    else if (d < d2) { d2 = d;  i2 = j; }
  }
  float w0 = 1.f / (fmaxf(d0, 0.f) + 1e-8f);
  float w1 = 1.f / (fmaxf(d1, 0.f) + 1e-8f);
  float w2 = 1.f / (fmaxf(d2, 0.f) + 1e-8f);
  float s = w0 + w1 + w2;
  idx3[t * 3 + 0] = i0; idx3[t * 3 + 1] = i1; idx3[t * 3 + 2] = i2;
  w3[t * 3 + 0] = w0 / s; w3[t * 3 + 1] = w1 / s; w3[t * 3 + 2] = w2 / s;
}

// FP input row: [interp(f2 via 3NN), f1, zero-pad]
__global__ void k_fp_input(const _Float16* __restrict__ f2, int ld2, int C2,
                           const _Float16* __restrict__ f1, int ld1, int C1,
                           const int* __restrict__ idx3, const float* __restrict__ w3,
                           _Float16* __restrict__ out, int ldo,
                           int n1, int n2, int total) {
  int t = blockIdx.x * blockDim.x + threadIdx.x;
  if (t >= total) return;
  int r = t / ldo, c = t - r * ldo;
  if (c < C2) {
    int b = r / n1;
    const _Float16* F = f2 + (size_t)b * n2 * ld2;
    float acc = 0.f;
#pragma unroll
    for (int k = 0; k < 3; ++k)
      acc += w3[r * 3 + k] * (float)F[(size_t)idx3[r * 3 + k] * ld2 + c];
    out[t] = (_Float16)acc;
  } else if (c < C2 + C1) {
    out[t] = f1[(size_t)r * ld1 + (c - C2)];
  } else {
    out[t] = (_Float16)0.f;
  }
}

__global__ void k_emit(const _Float16* __restrict__ h, int ld, float* __restrict__ out,
                       int C, int total) {
  int t = blockIdx.x * blockDim.x + threadIdx.x;
  if (t >= total) return;
  int r = t / C, c = t - r * C;
  out[t] = (float)h[(size_t)r * ld + c];
}

static inline int cdiv(int a, int b) { return (a + b - 1) / b; }

extern "C" void kernel_launch(void* const* d_in, const int* in_sizes, int n_in,
                              void* d_out, int out_size, void* d_ws, size_t ws_size,
                              hipStream_t stream) {
  (void)in_sizes; (void)n_in; (void)out_size; (void)ws_size;
  const int B = 4;
  const int np[5]   = {4096, 2048, 512, 128, 32};
  const int R[5]    = {B * 4096, B * 2048, B * 512, B * 128, B * 32};
  const float radii[5] = {0.1f, 0.2f, 0.4f, 0.8f, 1.6f};
  const int nsmp[5] = {32, 32, 32, 32, 16};

  char* base = (char*)d_ws; size_t off = 0;
  auto alloc = [&](size_t bytes) -> void* {
    void* p = base + off; off = (off + bytes + 255) & ~(size_t)255; return p;
  };

  float* xyz[5];
  for (int l = 0; l < 5; ++l) xyz[l] = (float*)alloc((size_t)R[l] * 3 * sizeof(float));
  _Float16* feats0 = (_Float16*)alloc((size_t)R[0] * 32 * sizeof(_Float16));
  const int lfC[5] = {32, 64, 128, 256, 512};
  _Float16* lf[5];
  for (int l = 0; l < 5; ++l) lf[l] = (_Float16*)alloc((size_t)R[l] * lfC[l] * sizeof(_Float16));
  const int nfC[4] = {64, 64, 128, 256};   // FP outputs for levels 0..3
  _Float16* nf[4];
  for (int l = 0; l < 4; ++l) nf[l] = (_Float16*)alloc((size_t)R[l] * nfC[l] * sizeof(_Float16));
  int*   idx  = (int*)alloc((size_t)R[0] * 32 * sizeof(int));     // max 524288
  int*   idx3 = (int*)alloc((size_t)R[0] * 3 * sizeof(int));
  float* w3   = (float*)alloc((size_t)R[0] * 3 * sizeof(float));
  const size_t bufElems = (size_t)R[0] * 32 * 32;                 // 16.8M halfs
  _Float16* bufA = (_Float16*)alloc(bufElems * sizeof(_Float16));
  _Float16* bufB = (_Float16*)alloc(bufElems * sizeof(_Float16));

  // 20 dense layers in d_in order: W at 1,3,...,39; b at 2,4,...,40.
  const int LK[20] = {3,32, 32,64, 64,128, 128,256, 256,512,
                      96,64, 192,128, 384,256, 768,512, 64,64};
  const int LN[20] = {32,32, 64,64, 128,128, 256,256, 512,512,
                      64,64, 128,64, 256,128, 512,256, 64,21};
  _Float16* Wp[20]; const float* Bb[20]; int Kp[20], Np[20];
  for (int i = 0; i < 20; ++i) {
    Kp[i] = (LK[i] + 31) & ~31;
    Np[i] = (LN[i] + 31) & ~31;
    Wp[i] = (_Float16*)alloc((size_t)Kp[i] * Np[i] * sizeof(_Float16));
    Bb[i] = (const float*)d_in[2 + 2 * i];
    int total = Kp[i] * Np[i];
    k_pack_w<<<cdiv(total, 256), 256, 0, stream>>>(
        (const float*)d_in[1 + 2 * i], Wp[i], LK[i], LN[i], Np[i] >> 4, total);
  }

  k_split_pc<<<cdiv(R[0], 256), 256, 0, stream>>>((const float*)d_in[0], xyz[0], feats0, R[0]);

  auto gemm = [&](const _Float16* A, int lda, int li, _Float16* out, int rows, int relu) {
    int waves = (rows / 16) * (Np[li] / 16);
    k_gemm<<<cdiv(waves * 32, 256), 256, 0, stream>>>(
        A, lda, Wp[li], Bb[li], LN[li], out, Np[li], rows, Kp[li], Np[li], relu);
  };

  auto sa_stage = [&](const float* xin, const float* ctr, int Nin, int M, int nsample,
                      float r, const _Float16* fin, int ldin, int li1, int li2,
                      _Float16* fout) {
    int nctr = B * M;
    k_ball_query<<<cdiv(nctr, 128), 128, 0, stream>>>(xin, ctr, idx, nctr, M, Nin,
                                                      nsample, r * r);
    int rowsOut = nctr * nsample;
    int total = rowsOut * ldin;
    k_gather<<<cdiv(total, 256), 256, 0, stream>>>(fin, idx, bufA, ldin,
                                                   M * nsample, Nin, total);
    gemm(bufA, ldin, li1, bufB, rowsOut, 1);
    gemm(bufB, Np[li1], li2, bufA, rowsOut, 1);
    int tp = nctr * Np[li2];
    k_maxpool<<<cdiv(tp, 256), 256, 0, stream>>>(bufA, fout, Np[li2], LN[li2], nsample, tp);
  };

  // conv0: centers = all points
  sa_stage(xyz[0], xyz[0], np[0], np[0], nsmp[0], radii[0], feats0, 32, 0, 1, lf[0]);
  // sa1..sa4 with strided centers
  for (int k = 1; k <= 4; ++k) {
    int stride = np[k - 1] / np[k];
    k_stride_centers<<<cdiv(R[k], 256), 256, 0, stream>>>(xyz[k - 1], xyz[k], R[k],
                                                          np[k], np[k - 1], stride);
    sa_stage(xyz[k - 1], xyz[k], np[k - 1], np[k], nsmp[k], radii[k],
             lf[k - 1], lfC[k - 1], 2 * k, 2 * k + 1, lf[k]);
  }

  auto fp_stage = [&](const float* x1, const float* x2, int n1, int n2,
                      const _Float16* f2, int ld2, int C2,
                      const _Float16* f1, int ld1, int C1,
                      int li1, int li2, _Float16* fout) {
    int n1all = B * n1;
    k_knn3<<<cdiv(n1all, 128), 128, 0, stream>>>(x1, x2, idx3, w3, n1all, n1, n2);
    int ldo = Kp[li1];
    int total = n1all * ldo;
    k_fp_input<<<cdiv(total, 256), 256, 0, stream>>>(f2, ld2, C2, f1, ld1, C1,
                                                     idx3, w3, bufA, ldo, n1, n2, total);
    gemm(bufA, ldo, li1, bufB, n1all, 1);
    gemm(bufB, Np[li1], li2, fout, n1all, 1);
  };

  // FP in reference order: deepest pair first, using freshly updated feats
  fp_stage(xyz[3], xyz[4], np[3], np[4], lf[4], 512, 512, lf[3], 256, 256, 16, 17, nf[3]);
  fp_stage(xyz[2], xyz[3], np[2], np[3], nf[3], 256, 256, lf[2], 128, 128, 14, 15, nf[2]);
  fp_stage(xyz[1], xyz[2], np[1], np[2], nf[2], 128, 128, lf[1],  64,  64, 12, 13, nf[1]);
  fp_stage(xyz[0], xyz[1], np[0], np[1], nf[1],  64,  64, lf[0],  32,  32, 10, 11, nf[0]);

  // Classifier: 64 -> 64 (relu) -> 21 (no relu), emit f32
  gemm(nf[0], 64, 18, bufA, R[0], 1);
  gemm(bufA, Np[18], 19, bufB, R[0], 0);
  int total = R[0] * 21;
  k_emit<<<cdiv(total, 256), 256, 0, stream>>>(bufB, Np[19], (float*)d_out, 21, total);
}
